// ReinforcementSummarization_88175678587295
// MI455X (gfx1250) — matrix operand order repaired and checked by reading, not compiled, not run on hardware
//
#include <hip/hip_runtime.h>

// ---------------------------------------------------------------------------
// Pointer-generator summarizer for MI455X (gfx1250), wave32 + WMMA bf16.
// All GEMMs are M=16 (batch) strips using V_WMMA_F32_16X16X32_BF16.
// vocab_W is converted to bf16 once (153.6MB < 192MB L2) so the 64 decoder
// steps re-read it from L2 instead of HBM. The vocab GEMM is N-blocked
// (4 tiles/wave) to amortize A-fragment loads.
// ---------------------------------------------------------------------------

#define SS 400
#define BB 16
#define TT 64
#define HH 512
#define EE 256
#define VV 50000
#define OOV 50
#define H2 256

typedef __attribute__((ext_vector_type(16))) __bf16 v16bf;
typedef __attribute__((ext_vector_type(8)))  __bf16 v8bf;
typedef __attribute__((ext_vector_type(8)))  float  v8f;

union V16BF { v16bf v; v8bf h[2]; __bf16 e[16]; };

__device__ __forceinline__ __bf16 f2bf(float f) {
  union { float f; unsigned u; } x; x.f = f;
  unsigned r = x.u + 0x7FFFu + ((x.u >> 16) & 1u);
  unsigned short hs = (unsigned short)(r >> 16);
  return __builtin_bit_cast(__bf16, hs);
}
__device__ __forceinline__ float bf2f(__bf16 h) {
  unsigned short s = __builtin_bit_cast(unsigned short, h);
  union { unsigned u; float f; } x; x.u = ((unsigned)s) << 16;
  return x.f;
}
__device__ __forceinline__ float sigf(float x) { return 1.f / (1.f + __expf(-x)); }

// ----------------------------- utility kernels -----------------------------

__global__ void k_cvt(const float* __restrict__ in, __bf16* __restrict__ out, int n) {
  int i = blockIdx.x * blockDim.x + threadIdx.x;
  if (i < n) out[i] = f2bf(in[i]);
}

__global__ void k_addbias(const float* __restrict__ a, const float* __restrict__ b,
                          float* __restrict__ out, int n) {
  int i = blockIdx.x * blockDim.x + threadIdx.x;
  if (i < n) out[i] = a[i] + b[i];
}

__global__ void k_gather_embed(const int* __restrict__ ids, const float* __restrict__ emb,
                               __bf16* __restrict__ out, int ntok) {
  int gid = blockIdx.x * blockDim.x + threadIdx.x;
  if (gid >= ntok * EE) return;
  int tk = gid / EE, e = gid % EE;
  out[gid] = f2bf(emb[(size_t)ids[tk] * EE + e]);
}

// ------------------------------ generic GEMM -------------------------------
// C[batch][16][N] = A[batch][16][K] @ W[N][K]^T  (+bias[N]) (+add[batch][16][N])
// One wave per 16-column tile; 8 waves (256 threads) per block.
__global__ void k_gemm_m16(const __bf16* __restrict__ A, const __bf16* __restrict__ W,
                           const float* __restrict__ bias, const float* __restrict__ add,
                           float* __restrict__ C, int N, int K) {
  const int wave = threadIdx.x >> 5;
  const int lane = threadIdx.x & 31;
  const int tile = blockIdx.x * 8 + wave;
  if (tile * 16 >= N) return;                 // wave-uniform
  const int n0 = tile * 16;
  const int batch = blockIdx.y;
  const int m  = lane & 15;
  const int hl = lane >> 4;

  const __bf16* arow = A + ((size_t)batch * 16 + m) * K + hl * 16;
  const __bf16* brow = W + (size_t)(n0 + m) * K + hl * 16;

  v8f acc = {};
  for (int k0 = 0; k0 < K; k0 += 32) {
    V16BF a, b;
    a.h[0] = *(const v8bf*)(arow + k0);
    a.h[1] = *(const v8bf*)(arow + k0 + 8);
    b.v    = *(const v16bf*)(brow + k0);
    acc = __builtin_amdgcn_wmma_f32_16x16x32_bf16(false, a.v, false, b.v,
                                                  (short)0, acc, false, false);
  }
  const float bn = bias ? bias[n0 + m] : 0.f;
  float* Cb = C + (size_t)batch * 16 * N;
  #pragma unroll
  for (int r = 0; r < 8; ++r) {
    int mm = r + hl * 8;
    float v = acc[r] + bn;
    if (add) v += add[((size_t)batch * 16 + mm) * N + n0 + m];
    Cb[(size_t)mm * N + n0 + m] = v;
  }
}

// ----------------------- N-blocked GEMM (vocab path) -----------------------
// Each wave computes 4 adjacent 16-col tiles (64 columns): A fragments loaded
// once per K-step and reused for 4 WMMAs (4 wmma / 10 b128-loads per step).
// Tail tiles (N % 64) fall back to the single-tile loop (wave-uniform branch).
__global__ void k_gemm_m16_n4(const __bf16* __restrict__ A, const __bf16* __restrict__ W,
                              const float* __restrict__ bias, float* __restrict__ C,
                              int N, int K) {
  const int wave = threadIdx.x >> 5;
  const int lane = threadIdx.x & 31;
  const int quad = blockIdx.x * 8 + wave;
  const int n0 = quad * 64;
  if (n0 >= N) return;                         // wave-uniform
  const int m  = lane & 15;
  const int hl = lane >> 4;
  const __bf16* arow = A + (size_t)m * K + hl * 16;

  if (n0 + 64 <= N) {                          // wave-uniform: full 4-tile path
    const __bf16* brow = W + (size_t)(n0 + m) * K + hl * 16;
    const size_t tstride = (size_t)16 * K;
    v8f acc0 = {}, acc1 = {}, acc2 = {}, acc3 = {};
    for (int k0 = 0; k0 < K; k0 += 32) {
      V16BF a, b0, b1, b2, b3;
      a.h[0] = *(const v8bf*)(arow + k0);
      a.h[1] = *(const v8bf*)(arow + k0 + 8);
      b0.v = *(const v16bf*)(brow + k0);
      b1.v = *(const v16bf*)(brow + tstride + k0);
      b2.v = *(const v16bf*)(brow + 2 * tstride + k0);
      b3.v = *(const v16bf*)(brow + 3 * tstride + k0);
      acc0 = __builtin_amdgcn_wmma_f32_16x16x32_bf16(false, a.v, false, b0.v,
                                                     (short)0, acc0, false, false);
      acc1 = __builtin_amdgcn_wmma_f32_16x16x32_bf16(false, a.v, false, b1.v,
                                                     (short)0, acc1, false, false);
      acc2 = __builtin_amdgcn_wmma_f32_16x16x32_bf16(false, a.v, false, b2.v,
                                                     (short)0, acc2, false, false);
      acc3 = __builtin_amdgcn_wmma_f32_16x16x32_bf16(false, a.v, false, b3.v,
                                                     (short)0, acc3, false, false);
    }
    #pragma unroll
    for (int q = 0; q < 4; ++q) {
      v8f acc = (q == 0) ? acc0 : (q == 1) ? acc1 : (q == 2) ? acc2 : acc3;
      const int nq = n0 + q * 16;
      const float bn = bias ? bias[nq + m] : 0.f;
      #pragma unroll
      for (int r = 0; r < 8; ++r) {
        int mm = r + hl * 8;
        C[(size_t)mm * N + nq + m] = acc[r] + bn;
      }
    }
  } else {                                     // wave-uniform: tail tiles
    for (int nq = n0; nq + 16 <= N; nq += 16) {
      const __bf16* brow = W + (size_t)(nq + m) * K + hl * 16;
      v8f acc = {};
      for (int k0 = 0; k0 < K; k0 += 32) {
        V16BF a, b;
        a.h[0] = *(const v8bf*)(arow + k0);
        a.h[1] = *(const v8bf*)(arow + k0 + 8);
        b.v    = *(const v16bf*)(brow + k0);
        acc = __builtin_amdgcn_wmma_f32_16x16x32_bf16(false, a.v, false, b.v,
                                                      (short)0, acc, false, false);
      }
      const float bn = bias ? bias[nq + m] : 0.f;
      #pragma unroll
      for (int r = 0; r < 8; ++r) {
        int mm = r + hl * 8;
        C[(size_t)mm * N + nq + m] = acc[r] + bn;
      }
    }
  }
}

// --------------------------- persistent encoder ----------------------------
// grid.x = 2 (dir 0 = fwd, 1 = bwd). h/c live in LDS; recurrent matmul = WMMA.
__global__ void k_lstm_encoder(const float* __restrict__ xproj,   // [2][S][16][1024]
                               const __bf16* __restrict__ Whh,    // [2][1024][256]
                               float* __restrict__ gates,         // [2][16][1024] scratch
                               float* __restrict__ enc_seq,       // [S][16][512]
                               float* __restrict__ hfin,          // [2][16][256]
                               float* __restrict__ cfin) {        // [2][16][256]
  const int dir  = blockIdx.x;
  const int tid  = threadIdx.x;
  const int wave = tid >> 5, lane = tid & 31;
  const int m = lane & 15, hl = lane >> 4;

  __shared__ __bf16 hS[16 * 256];
  __shared__ float  cS[16 * 256];
  for (int e = tid; e < 4096; e += 256) { hS[e] = f2bf(0.f); cS[e] = 0.f; }
  __syncthreads();

  const __bf16* Wd = Whh + (size_t)dir * 1024 * 256;
  float* gd = gates + (size_t)dir * 16 * 1024;

  for (int step = 0; step < SS; ++step) {
    const int s = dir ? (SS - 1 - step) : step;
    const float* xp = xproj + ((size_t)dir * SS + s) * 16 * 1024;

    for (int tile = wave; tile < 64; tile += 8) {
      const int n0 = tile * 16;
      const __bf16* brow = Wd + (size_t)(n0 + m) * 256 + hl * 16;
      v8f acc = {};
      for (int k0 = 0; k0 < 256; k0 += 32) {
        V16BF a, b;
        const int ab = m * 256 + k0 + hl * 16;
        #pragma unroll
        for (int u = 0; u < 8; ++u) { a.e[u] = hS[ab + u]; a.e[8 + u] = hS[ab + 8 + u]; }
        b.v = *(const v16bf*)(brow + k0);
        acc = __builtin_amdgcn_wmma_f32_16x16x32_bf16(false, a.v, false, b.v,
                                                      (short)0, acc, false, false);
      }
      #pragma unroll
      for (int r = 0; r < 8; ++r) {
        int mm = r + hl * 8;
        gd[mm * 1024 + n0 + m] = acc[r] + xp[mm * 1024 + n0 + m];
      }
    }
    __syncthreads();
    for (int e = tid; e < 4096; e += 256) {
      int mm = e >> 8, j = e & 255;
      float gi = gd[mm * 1024 + j];
      float gf = gd[mm * 1024 + 256 + j];
      float gg = gd[mm * 1024 + 512 + j];
      float go = gd[mm * 1024 + 768 + j];
      float c  = sigf(gf) * cS[e] + sigf(gi) * tanhf(gg);
      float h  = sigf(go) * tanhf(c);
      cS[e] = c; hS[e] = f2bf(h);
      enc_seq[((size_t)s * 16 + mm) * 512 + dir * 256 + j] = h;
    }
    __syncthreads();
  }
  for (int e = tid; e < 4096; e += 256) {
    hfin[dir * 4096 + e] = bf2f(hS[e]);
    cfin[dir * 4096 + e] = cS[e];
  }
}

// h0 = stack([hf,hb]).reshape(-1,H)  (faithful batch-mixing reshape)
__global__ void k_h0c0(const float* __restrict__ hfin, const float* __restrict__ cfin,
                       float* __restrict__ hcur, float* __restrict__ ccur,
                       __bf16* __restrict__ h16) {
  int gid = blockIdx.x * blockDim.x + threadIdx.x;
  if (gid >= 16 * 512) return;
  int r = gid >> 9, col = gid & 511;
  int d = r >> 3;
  int b = ((r & 7) << 1) + (col >> 8);
  int j = col & 255;
  float hv = hfin[d * 4096 + b * 256 + j];
  float cv = cfin[d * 4096 + b * 256 + j];
  hcur[gid] = hv; ccur[gid] = cv; h16[gid] = f2bf(hv);
}

__global__ void k_lstm_update_dec(const float* __restrict__ gates,
                                  float* __restrict__ hcur, float* __restrict__ ccur,
                                  __bf16* __restrict__ h16, float* __restrict__ hid_buf,
                                  int t) {
  int e = blockIdx.x * blockDim.x + threadIdx.x;
  if (e >= 16 * 512) return;
  int m = e >> 9, j = e & 511;
  const float* g = gates + (size_t)m * 2048;
  float gi = g[j], gf = g[512 + j], gg = g[1024 + j], go = g[1536 + j];
  float c = sigf(gf) * ccur[e] + sigf(gi) * tanhf(gg);
  float h = sigf(go) * tanhf(c);
  ccur[e] = c; hcur[e] = h; h16[e] = f2bf(h);
  hid_buf[(size_t)t * 16 * 512 + e] = h;
}

// intra-temporal attention over encoder states (one block per batch row)
__global__ void k_enc_attention(const float* __restrict__ hcur,
                                const float* __restrict__ featproj,  // [S][16][512]
                                float* __restrict__ attnbuf,         // [16][S]
                                float* __restrict__ enc_ctx,         // [16][512]
                                int t) {
  const int b = blockIdx.x, tid = threadIdx.x;
  __shared__ float hS[512];
  __shared__ float sc[SS];
  __shared__ float red[256];
  for (int j = tid; j < 512; j += 256) hS[j] = hcur[b * 512 + j];
  __syncthreads();
  float part = 0.f;
  for (int s = tid; s < SS; s += 256) {
    const float* fp = featproj + ((size_t)s * 16 + b) * 512;
    float dot = 0.f;
    for (int j = 0; j < 512; ++j) dot += hS[j] * fp[j];
    float e = __expf(dot);
    if (t > 0) e /= attnbuf[b * SS + s];
    sc[s] = e; part += e;
  }
  red[tid] = part;
  __syncthreads();
  for (int off = 128; off > 0; off >>= 1) {
    if (tid < off) red[tid] += red[tid + off];
    __syncthreads();
  }
  const float inv = 1.f / red[0];
  for (int s = tid; s < SS; s += 256) {
    float a = sc[s] * inv;
    sc[s] = a; attnbuf[b * SS + s] = a;
  }
  __syncthreads();
  for (int j = tid; j < 512; j += 256) {
    float acc = 0.f;
    for (int s = 0; s < SS; ++s) acc += sc[s] * featproj[((size_t)s * 16 + b) * 512 + j];
    enc_ctx[b * 512 + j] = acc;
  }
}

// intra-decoder attention + feats assembly + pointer gate p
__global__ void k_dec_attention(const float* __restrict__ hcur,
                                const float* __restrict__ keys,     // [T][16][512]
                                const float* __restrict__ hid_buf,  // [T][16][512]
                                const float* __restrict__ enc_ctx,
                                const float* __restrict__ ptrW, const float* __restrict__ ptrb,
                                float* __restrict__ feats32, __bf16* __restrict__ feats16,
                                float* __restrict__ pbuf, int t) {
  const int b = blockIdx.x, tid = threadIdx.x;
  __shared__ float hS[512];
  __shared__ float a[TT];
  __shared__ float red[256];
  for (int j = tid; j < 512; j += 256) hS[j] = hcur[b * 512 + j];
  __syncthreads();
  if (tid < TT) {
    float v = -1e30f;
    if (tid < t) {
      const float* kp = keys + ((size_t)tid * 16 + b) * 512;
      float dot = 0.f;
      for (int j = 0; j < 512; ++j) dot += hS[j] * kp[j];
      v = dot;
    }
    a[tid] = v;
  }
  __syncthreads();
  if (tid == 0) {
    float mx = a[0];
    for (int i = 1; i < TT; ++i) mx = fmaxf(mx, a[i]);
    float sum = 0.f;
    for (int i = 0; i < TT; ++i) { float e = __expf(a[i] - mx); a[i] = e; sum += e; }
    float inv = 1.f / sum;
    for (int i = 0; i < TT; ++i) a[i] *= inv;
  }
  __syncthreads();
  for (int j = tid; j < 512; j += 256) {
    float d = 0.f;
    if (t > 0)
      for (int tp = 0; tp < t; ++tp) d += a[tp] * hid_buf[((size_t)tp * 16 + b) * 512 + j];
    float hv = hS[j], ev = enc_ctx[b * 512 + j];
    size_t base = (size_t)b * 1536;
    feats32[base + j]        = hv; feats16[base + j]        = f2bf(hv);
    feats32[base + 512 + j]  = ev; feats16[base + 512 + j]  = f2bf(ev);
    feats32[base + 1024 + j] = d;  feats16[base + 1024 + j] = f2bf(d);
  }
  __syncthreads();
  float part = 0.f;
  for (int i = tid; i < 1536; i += 256) part += feats32[(size_t)b * 1536 + i] * ptrW[i];
  red[tid] = part;
  __syncthreads();
  for (int off = 128; off > 0; off >>= 1) {
    if (tid < off) red[tid] += red[tid + off];
    __syncthreads();
  }
  if (tid == 0) pbuf[b] = sigf(red[0] + ptrb[0]);
}

__global__ void k_row_reduce(const float* __restrict__ logits,
                             float* __restrict__ rowmax, float* __restrict__ rowinv) {
  const int b = blockIdx.x, tid = threadIdx.x;
  __shared__ float red[256];
  float mx = -1e30f;
  for (int v = tid; v < VV; v += 256) mx = fmaxf(mx, logits[(size_t)b * VV + v]);
  red[tid] = mx;
  __syncthreads();
  for (int off = 128; off > 0; off >>= 1) {
    if (tid < off) red[tid] = fmaxf(red[tid], red[tid + off]);
    __syncthreads();
  }
  const float M = red[0];
  __syncthreads();
  float s = 0.f;
  for (int v = tid; v < VV; v += 256) s += __expf(logits[(size_t)b * VV + v] - M);
  red[tid] = s;
  __syncthreads();
  for (int off = 128; off > 0; off >>= 1) {
    if (tid < off) red[tid] += red[tid + off];
    __syncthreads();
  }
  if (tid == 0) { rowmax[b] = M; rowinv[b] = 1.f / red[0]; }
}

__global__ void k_finalize(const float* __restrict__ logits,
                           const float* __restrict__ rowmax, const float* __restrict__ rowinv,
                           const float* __restrict__ pbuf, float* __restrict__ out, int t) {
  int gid = blockIdx.x * blockDim.x + threadIdx.x;
  const int VE = VV + OOV;
  if (gid >= 16 * VE) return;
  int b = gid / VE, v = gid % VE;
  float val = 0.f;
  if (v < VV)
    val = (1.f - pbuf[b]) * __expf(logits[(size_t)b * VV + v] - rowmax[b]) * rowinv[b];
  out[((size_t)t * 16 + b) * VE + v] = val;
}

__global__ void k_scatter(const int* __restrict__ inputs_ext,   // [S][16]
                          const float* __restrict__ attnbuf, const float* __restrict__ pbuf,
                          float* __restrict__ out, int t) {
  int gid = blockIdx.x * blockDim.x + threadIdx.x;
  if (gid >= 16 * SS) return;
  int b = gid / SS, s = gid % SS;
  int id = inputs_ext[s * 16 + b];
  atomicAdd(&out[((size_t)t * 16 + b) * (VV + OOV) + id], pbuf[b] * attnbuf[b * SS + s]);
}

// ------------------------------- host driver -------------------------------

extern "C" void kernel_launch(void* const* d_in, const int* in_sizes, int n_in,
                              void* d_out, int out_size, void* d_ws, size_t ws_size,
                              hipStream_t stream) {
  (void)in_sizes; (void)n_in; (void)out_size; (void)ws_size;
  const int*   inputs     = (const int*)d_in[0];
  const int*   inputs_ext = (const int*)d_in[2];
  const int*   outputs    = (const int*)d_in[4];
  const float* embedding  = (const float*)d_in[5];
  const float* enc_Wih_f  = (const float*)d_in[6];
  const float* enc_Whh_f  = (const float*)d_in[7];
  const float* enc_bih_f  = (const float*)d_in[8];
  const float* enc_bhh_f  = (const float*)d_in[9];
  const float* enc_Wih_b  = (const float*)d_in[10];
  const float* enc_Whh_b  = (const float*)d_in[11];
  const float* enc_bih_b  = (const float*)d_in[12];
  const float* enc_bhh_b  = (const float*)d_in[13];
  const float* feat_W     = (const float*)d_in[14];
  const float* dec_Wih    = (const float*)d_in[15];
  const float* dec_Whh    = (const float*)d_in[16];
  const float* dec_bih    = (const float*)d_in[17];
  const float* dec_bhh    = (const float*)d_in[18];
  const float* attn_W     = (const float*)d_in[19];
  const float* vocab_W    = (const float*)d_in[20];
  const float* vocab_b    = (const float*)d_in[21];
  const float* ptr_W      = (const float*)d_in[22];
  const float* ptr_b      = (const float*)d_in[23];
  float* out = (float*)d_out;

  const int G2 = 4 * H2;   // 1024
  const int G4 = 4 * HH;   // 2048

  // ---- workspace carve (256B aligned bump allocator) ----
  char* wp = (char*)d_ws;
  auto alloc = [&](size_t bytes) -> void* {
    void* p = (void*)wp;
    wp += (bytes + 255) & ~(size_t)255;
    return p;
  };
  __bf16* WihF16   = (__bf16*)alloc((size_t)G2 * EE * 2);
  __bf16* WihB16   = (__bf16*)alloc((size_t)G2 * EE * 2);
  __bf16* Whh16    = (__bf16*)alloc((size_t)2 * G2 * H2 * 2);     // [2][1024][256]
  __bf16* decWih16 = (__bf16*)alloc((size_t)G4 * EE * 2);
  __bf16* decWhh16 = (__bf16*)alloc((size_t)G4 * HH * 2);
  __bf16* attnW16  = (__bf16*)alloc((size_t)HH * HH * 2);
  __bf16* featW16  = (__bf16*)alloc((size_t)HH * HH * 2);
  __bf16* vocabW16 = (__bf16*)alloc((size_t)VV * 3 * HH * 2);     // 153.6 MB (L2-resident)
  __bf16* x16      = (__bf16*)alloc((size_t)SS * BB * EE * 2);
  __bf16* yemb16   = (__bf16*)alloc((size_t)TT * BB * EE * 2);
  __bf16* encseq16 = (__bf16*)alloc((size_t)SS * BB * HH * 2);
  __bf16* h16      = (__bf16*)alloc((size_t)BB * HH * 2);
  __bf16* feats16  = (__bf16*)alloc((size_t)BB * 3 * HH * 2);

  float* biasF    = (float*)alloc((size_t)G2 * 4);
  float* biasB    = (float*)alloc((size_t)G2 * 4);
  float* biasD    = (float*)alloc((size_t)G4 * 4);
  float* xprojEnc = (float*)alloc((size_t)2 * SS * 16 * G2 * 4);  // [2][S][16][1024]
  float* gatesEnc = (float*)alloc((size_t)2 * 16 * G2 * 4);
  float* enc_seq  = (float*)alloc((size_t)SS * 16 * HH * 4);
  float* hfin     = (float*)alloc((size_t)2 * 16 * H2 * 4);
  float* cfin     = (float*)alloc((size_t)2 * 16 * H2 * 4);
  float* featproj = (float*)alloc((size_t)SS * 16 * HH * 4);
  float* hcur     = (float*)alloc((size_t)16 * HH * 4);
  float* ccur     = (float*)alloc((size_t)16 * HH * 4);
  float* xprojDec = (float*)alloc((size_t)TT * 16 * G4 * 4);
  float* gatesD   = (float*)alloc((size_t)16 * G4 * 4);
  float* keys     = (float*)alloc((size_t)TT * 16 * HH * 4);
  float* hid_buf  = (float*)alloc((size_t)TT * 16 * HH * 4);
  float* enc_ctx  = (float*)alloc((size_t)16 * HH * 4);
  float* attnbuf  = (float*)alloc((size_t)16 * SS * 4);
  float* feats32  = (float*)alloc((size_t)16 * 3 * HH * 4);
  float* logits   = (float*)alloc((size_t)16 * VV * 4);
  float* pbuf     = (float*)alloc(16 * 4);
  float* rowmax   = (float*)alloc(16 * 4);
  float* rowinv   = (float*)alloc(16 * 4);

  auto cvt = [&](const float* src, __bf16* dst, int n) {
    k_cvt<<<(n + 255) / 256, 256, 0, stream>>>(src, dst, n);
  };

  // ---- weight conversions (once per launch) ----
  cvt(enc_Wih_f, WihF16, G2 * EE);
  cvt(enc_Wih_b, WihB16, G2 * EE);
  cvt(enc_Whh_f, Whh16, G2 * H2);
  cvt(enc_Whh_b, Whh16 + (size_t)G2 * H2, G2 * H2);
  cvt(dec_Wih, decWih16, G4 * EE);
  cvt(dec_Whh, decWhh16, G4 * HH);
  cvt(attn_W, attnW16, HH * HH);
  cvt(feat_W, featW16, HH * HH);
  cvt(vocab_W, vocabW16, VV * 3 * HH);
  k_addbias<<<4, 256, 0, stream>>>(enc_bih_f, enc_bhh_f, biasF, G2);
  k_addbias<<<4, 256, 0, stream>>>(enc_bih_b, enc_bhh_b, biasB, G2);
  k_addbias<<<8, 256, 0, stream>>>(dec_bih, dec_bhh, biasD, G4);

  // ---- embedding gathers ----
  k_gather_embed<<<(SS * BB * EE + 255) / 256, 256, 0, stream>>>(inputs, embedding, x16, SS * BB);
  k_gather_embed<<<(TT * BB * EE + 255) / 256, 256, 0, stream>>>(outputs, embedding, yemb16, TT * BB);

  // ---- encoder: batched input projections, then persistent recurrent loop ----
  k_gemm_m16<<<dim3(8, SS), 256, 0, stream>>>(x16, WihF16, biasF, nullptr, xprojEnc, G2, EE);
  k_gemm_m16<<<dim3(8, SS), 256, 0, stream>>>(x16, WihB16, biasB, nullptr,
                                              xprojEnc + (size_t)SS * 16 * G2, G2, EE);
  k_lstm_encoder<<<2, 256, 0, stream>>>(xprojEnc, Whh16, gatesEnc, enc_seq, hfin, cfin);

  // ---- feat projection: enc_out (accessed as featproj[s][b][h]) ----
  cvt(enc_seq, encseq16, SS * 16 * HH);
  k_gemm_m16<<<dim3(4, SS), 256, 0, stream>>>(encseq16, featW16, nullptr, nullptr, featproj, HH, HH);

  // ---- decoder init + input projections ----
  k_h0c0<<<32, 256, 0, stream>>>(hfin, cfin, hcur, ccur, h16);
  k_gemm_m16<<<dim3(16, TT), 256, 0, stream>>>(yemb16, decWih16, biasD, nullptr, xprojDec, G4, EE);

  // vocab GEMM grid: ceil(ceil(50000/64)/8) = ceil(782/8) = 98 blocks
  const int vquads  = (VV + 63) / 64;
  const int vblocks = (vquads + 7) / 8;

  // ---- decoder loop ----
  for (int t = 0; t < TT; ++t) {
    k_gemm_m16<<<dim3(16, 1), 256, 0, stream>>>(h16, decWhh16, nullptr,
                                                xprojDec + (size_t)t * 16 * G4, gatesD, G4, HH);
    k_lstm_update_dec<<<32, 256, 0, stream>>>(gatesD, hcur, ccur, h16, hid_buf, t);
    k_gemm_m16<<<dim3(4, 1), 256, 0, stream>>>(h16, attnW16, nullptr, nullptr,
                                               keys + (size_t)t * 16 * HH, HH, HH);
    k_enc_attention<<<16, 256, 0, stream>>>(hcur, featproj, attnbuf, enc_ctx, t);
    k_dec_attention<<<16, 256, 0, stream>>>(hcur, keys, hid_buf, enc_ctx, ptr_W, ptr_b,
                                            feats32, feats16, pbuf, t);
    // vocab projection: N-blocked, bf16 weights stay L2-resident across steps
    k_gemm_m16_n4<<<dim3(vblocks, 1), 256, 0, stream>>>(feats16, vocabW16, vocab_b,
                                                        logits, VV, 3 * HH);
    k_row_reduce<<<16, 256, 0, stream>>>(logits, rowmax, rowinv);
    k_finalize<<<(16 * (VV + OOV) + 255) / 256, 256, 0, stream>>>(logits, rowmax, rowinv,
                                                                  pbuf, out, t);
    k_scatter<<<(16 * SS + 255) / 256, 256, 0, stream>>>(inputs_ext, attnbuf, pbuf, out, t);
  }
}
